// Block_76501957476974
// MI455X (gfx1250) — compile-verified, matrix operand order checked by reading, not run on hardware
//
#include <hip/hip_runtime.h>
#include <math.h>

// ---------------------------------------------------------------------------
// Causal dilated conv1d (dilation=64, K=2, Cin=Cout=64) + tanh*sigmoid gate.
// y[b,o,t] = sum_i w[o,i,0]*x[b,i,t-64] + sum_i w[o,i,1]*x[b,i,t]
// f16 WMMA GEMM with f32 accumulation. Memory-bound: 4.3 GFLOP / 134 MB HBM
// -> needs ~745 TFLOP/s to stay on the 23.3 TB/s roof; only the WMMA pipe
// delivers that. Epilogue is branch-free (1 exp + 1 rcp per element).
// ---------------------------------------------------------------------------

typedef __attribute__((ext_vector_type(16))) _Float16 v16h;
typedef __attribute__((ext_vector_type(2)))  _Float16 v2h;
typedef __attribute__((ext_vector_type(8)))  float    v8f;

#define T_LEN   16384
#define NCH     64
#define NBATCH  16
#define BN      128          // output time-columns per block
#define HALO    64           // dilation * (K-1)
#define COLS    (BN + HALO)  // 192 LDS columns (halo on the left)

union FragU { v16h v; float4 q[2]; };

// Load one wave-lane's A/B fragment for v_wmma_f32_16x16x32_f16 from a
// row-major f16 LDS array with 64 halves per row.
// Per the CDNA5 ISA 16-bit 16x32 layout, each lane needs two contiguous
// 8-half (16B) runs: K = kfirst and K = kfirst+16  -> two ds_load_b128.
__device__ inline v16h load_frag(const _Float16* base, int row, int kfirst) {
  FragU f;
  const char* p = (const char*)(base + row * 64 + kfirst);
  f.q[0] = *(const float4*)(p);
  f.q[1] = *(const float4*)(p + 32);
  return f.v;
}

// Branch-free tanh(y)*sigmoid(y):
//   e = exp(-y); sigmoid = 1/(1+e); tanh = (1-e^2)/(1+e^2)  [e^2 = exp(-2y)]
//   gate = (1-e^2) * rcp((1+e^2)*(1+e))
__device__ inline float gated(float y) {
  y = fminf(fmaxf(y, -30.0f), 30.0f);   // keep e^2 finite (|y| ~ 1 here)
#if __has_builtin(__builtin_amdgcn_exp2f)
  const float e = __builtin_amdgcn_exp2f(-1.442695040888963f * y); // exp(-y)
#else
  const float e = __expf(-y);
#endif
  const float e2  = e * e;
  const float den = (1.0f + e2) * (1.0f + e);
#if __has_builtin(__builtin_amdgcn_rcpf)
  return (1.0f - e2) * __builtin_amdgcn_rcpf(den);
#else
  return (1.0f - e2) / den;
#endif
}

__global__ __launch_bounds__(256) void wavenet_conv_gate(
    const float* __restrict__ x,   // [B, 64, 16384]
    const float* __restrict__ w,   // [64, 64, 2]
    float* __restrict__ out)       // [B, 64, 16384]
{
  __shared__ __align__(16) _Float16 sW[2 * 64 * 64]; // [tap][o][i], 16 KB
  __shared__ __align__(16) _Float16 sX[COLS * 64];   // [col][ch],   24 KB

  const int b   = blockIdx.y;
  const int t0  = blockIdx.x * BN;
  const int tid = threadIdx.x;

  // ---- Stage weights: w[o][i][k] f32 -> sW[k][o][i] f16 (pairs along i) ----
  for (int p = tid; p < 4096; p += 256) {
    const int ipair = p & 31;          // i/2
    const int o     = (p >> 5) & 63;
    const int k     = p >> 11;         // tap
    const float* g  = w + o * 128 + ipair * 4 + k;   // w[o][2*ipair][k]
    v2h pk;
    pk.x = (_Float16)g[0];
    pk.y = (_Float16)g[2];                           // w[o][2*ipair+1][k]
    *(v2h*)&sW[k * 4096 + o * 64 + ipair * 2] = pk;
  }

  // ---- Stage x tile (with left halo): x[b][ch][t] -> sX[col][ch] f16 ----
  const float* xb = x + (size_t)b * NCH * T_LEN;
  for (int p = tid; p < COLS * 32; p += 256) {
    const int cp  = p & 31;            // ch/2
    const int col = p >> 5;            // 0..191
    const int t   = t0 - HALO + col;
    v2h pk;
    if (t >= 0) {
      pk.x = (_Float16)xb[(size_t)(cp * 2)     * T_LEN + t];
      pk.y = (_Float16)xb[(size_t)(cp * 2 + 1) * T_LEN + t];
    } else {
      pk.x = (_Float16)0.0f;           // causal zero-pad
      pk.y = (_Float16)0.0f;
    }
    *(v2h*)&sX[col * 64 + cp * 2] = pk;
  }
  __syncthreads();

  // ---- WMMA compute: 8 waves = 4 M-tiles x 2 N-groups ----
  const int lane  = tid & 31;
  const int wave  = tid >> 5;
  const int wm    = wave & 3;              // which 16-row block of C_out
  const int wn    = wave >> 2;             // which 64-column half of the tile
  const int lrow  = lane & 15;
  const int khalf = (lane >> 4) * 8;       // per-lane K-run select (0 or 8)

  // A fragments (weights), fixed per wave: [tap][kb]
  v16h A[2][2];
#pragma unroll
  for (int tap = 0; tap < 2; ++tap)
#pragma unroll
    for (int kb = 0; kb < 2; ++kb)
      A[tap][kb] = load_frag(sW + tap * 4096, wm * 16 + lrow, kb * 32 + khalf);

  float* ob = out + (size_t)b * NCH * T_LEN;

#pragma unroll
  for (int j = 0; j < 4; ++j) {
    const int n = wn * 64 + j * 16 + lrow;       // tile-relative output column
    // tap0 reads halo-shifted columns (t-64 -> LDS col n),
    // tap1 reads in-place columns    (t    -> LDS col n+64)
    v16h B00 = load_frag(sX, n,       0  + khalf);
    v16h B01 = load_frag(sX, n,       32 + khalf);
    v16h B10 = load_frag(sX, n + 64,  0  + khalf);
    v16h B11 = load_frag(sX, n + 64,  32 + khalf);

    v8f acc = {};
    acc = __builtin_amdgcn_wmma_f32_16x16x32_f16(false, A[0][0], false, B00,
                                                 (short)0, acc, false, false);
    acc = __builtin_amdgcn_wmma_f32_16x16x32_f16(false, A[0][1], false, B01,
                                                 (short)0, acc, false, false);
    acc = __builtin_amdgcn_wmma_f32_16x16x32_f16(false, A[1][0], false, B10,
                                                 (short)0, acc, false, false);
    acc = __builtin_amdgcn_wmma_f32_16x16x32_f16(false, A[1][1], false, B11,
                                                 (short)0, acc, false, false);

    // ---- gated activation + store (C layout: VGPR r -> M = r + 8*(lane>=16),
    //      N = lane&15). Rows are coalesced across lanes 0-15 / 16-31. ----
    const int tcol  = t0 + wn * 64 + j * 16 + lrow;
    const int obase = wm * 16 + (lane >> 4) * 8;
#pragma unroll
    for (int r = 0; r < 8; ++r) {
      float* dst = ob + (size_t)(obase + r) * T_LEN + tcol;
      __builtin_nontemporal_store(gated(acc[r]), dst);  // streamed, write-once
    }
  }
}

extern "C" void kernel_launch(void* const* d_in, const int* in_sizes, int n_in,
                              void* d_out, int out_size, void* d_ws, size_t ws_size,
                              hipStream_t stream) {
  const float* x = (const float*)d_in[0];
  const float* w = (const float*)d_in[1];
  float* out     = (float*)d_out;
  dim3 grid(T_LEN / BN, NBATCH);   // (128, 16) blocks
  hipLaunchKernelGGL(wavenet_conv_gate, grid, dim3(256), 0, stream, x, w, out);
}
